// WindowAttention3D_55044300866251
// MI455X (gfx1250) — matrix operand order repaired and checked by reading, not compiled, not run on hardware
//
#include <hip/hip_runtime.h>
#include <hip/hip_bf16.h>
#include <cstdint>

typedef __attribute__((ext_vector_type(16))) _Float16 v16h;
typedef __attribute__((ext_vector_type(8)))  float    v8f;

#define N_TOK 98
#define NP    112          // padded tokens (7 * 16)
#define NT_M  7
#define CDIM  128
#define NHEAD 4
#define HD    32
#define NWIN  64
#define XSTR  136          // f16 row stride (halves): 16B-aligned rows + conflict-free lane spread
#define VSTR  136
#define SSTR  116          // f32 row stride for scores
#define QSCALE 0.17677669529663687f   // 32^-0.5
#define NN    (N_TOK * N_TOK)         // 9604
#define NTHREADS 512                   // 16 waves -> 4 waves/SIMD at 1 WG/WGP
#define NWAVES   16

// ---- LDS byte offsets (dynamic shared) ----
#define XS_OFF 0                        // [112][136] f16 (aliased as PS after stage 1)
#define QS_OFF 30464
#define KS_OFF 60928
#define VT_OFF 91392                    // [128][136] f16, channel-major
#define SS_OFF 126208                   // [112][116] f32
#define OS_OFF 178176                   // [112][136] f16
#define LDS_BYTES 208640

// ---- workspace byte offsets ----
#define WQ_OFF  0                       // 384*128 f16
#define WP_OFF  98304                   // 128*128 f16
#define BH_OFF  131072                  // 4*98*98 f32 (bias_table gathered per head)

union FragU { v16h v; uint32_t u[8]; };

// A-matrix 16x32 f16 fragment (ISA 7.12.2): row-major buf[m][k], stride in halves.
__device__ __forceinline__ v16h load_a(const _Float16* buf, int stride, int mbase, int kbase) {
  int lane = threadIdx.x & 31;
  int m = mbase + (lane & 15);
  int hi = lane >> 4;
  FragU f;
#pragma unroll
  for (int j = 0; j < 8; ++j) {
    int k = (j < 4 ? 2 * j : 16 + 2 * (j - 4)) + hi * 8;
    f.u[j] = *(const uint32_t*)(buf + m * stride + kbase + k);
  }
  return f.v;
}

// B-matrix 32x16 f16 fragment, stored transposed buf[n][k] (K-pairs contiguous).
__device__ __forceinline__ v16h load_b(const _Float16* buf, int stride, int nbase, int kbase) {
  int lane = threadIdx.x & 31;
  int n = nbase + (lane & 15);
  int hi = lane >> 4;
  FragU f;
#pragma unroll
  for (int j = 0; j < 8; ++j) {
    int k = (j < 4 ? 2 * j : 16 + 2 * (j - 4)) + hi * 8;
    f.u[j] = *(const uint32_t*)(buf + n * stride + kbase + k);
  }
  return f.v;
}

__device__ __forceinline__ v8f wmma_f16(v16h a, v16h b, v8f c) {
  return __builtin_amdgcn_wmma_f32_16x16x32_f16(false, a, false, b, (short)0, c, false, false);
}

// fp32 weights -> f16 staging in workspace (once per launch, L2-resident after)
__global__ void prep_weights(const float* __restrict__ qkv_w, const float* __restrict__ proj_w,
                             _Float16* __restrict__ wq, _Float16* __restrict__ wp) {
  int i = blockIdx.x * 256 + threadIdx.x;
  if (i < 3 * CDIM * CDIM) wq[i] = (_Float16)qkv_w[i];
  if (i < CDIM * CDIM)     wp[i] = (_Float16)proj_w[i];
}

// gather bias_table[rel_index] into dense per-head planes: biasH[h][m*98+n]
__global__ void prep_bias(const float* __restrict__ bias_table, const int* __restrict__ rel_index,
                          float* __restrict__ biasH) {
  int i = blockIdx.x * 256 + threadIdx.x;
  if (i < NHEAD * NN) {
    int h = i / NN, e = i % NN;
    biasH[i] = bias_table[rel_index[e] * NHEAD + h];
  }
}

__global__ __launch_bounds__(NTHREADS, 1)
void win_attn3d_kernel(const float* __restrict__ x,
                       const float* __restrict__ attn_mask,
                       const float* __restrict__ qkv_b,
                       const float* __restrict__ proj_b,
                       const float* __restrict__ biasH,
                       const _Float16* __restrict__ wq,
                       const _Float16* __restrict__ wp,
                       float* __restrict__ out) {
  extern __shared__ __align__(16) char smem[];
  _Float16* xs = (_Float16*)(smem + XS_OFF);
  _Float16* qs = (_Float16*)(smem + QS_OFF);
  _Float16* ks = (_Float16*)(smem + KS_OFF);
  _Float16* vt = (_Float16*)(smem + VT_OFF);
  float*    ss = (float*)   (smem + SS_OFF);
  _Float16* os = (_Float16*)(smem + OS_OFF);
  _Float16* ps = xs;   // probs alias x stage buffer

  const int b   = blockIdx.x;
  const int tid = threadIdx.x;
  const int wv  = tid >> 5;
  const int ln  = tid & 31;
  const float* maskw = attn_mask + (size_t)(b & (NWIN - 1)) * NN;

  // ---- stage 0: stage x (fp32->f16), zero row padding; zero vt; prefetch mask ----
  const float* xb = x + (size_t)b * N_TOK * CDIM;
  for (int i = tid; i < NP * CDIM; i += NTHREADS) {
    int r = i >> 7, c = i & 127;
    float v = (r < N_TOK) ? xb[r * CDIM + c] : 0.f;
    xs[r * XSTR + c] = (_Float16)v;
  }
  for (int i = tid; i < CDIM * VSTR; i += NTHREADS) vt[i] = (_Float16)0.f;
  for (int i = tid * 32; i < NN; i += NTHREADS * 32) __builtin_prefetch(maskw + i, 0, 2);
  __syncthreads();

  // ---- stage 1: QKV projection, 7x24 output tiles, K = 4 x 32 ----
  for (int t = wv; t < NT_M * 24; t += NWAVES) {
    int mt = t % NT_M, jt = t / NT_M;           // jt uniform per tile -> uniform branch below
    int mb = mt * 16, j0 = jt * 16;
    v8f c = {};
#pragma unroll
    for (int kc = 0; kc < 4; ++kc) {
      v16h a  = load_a(xs, XSTR, mb, kc * 32);
      v16h bf = load_b(wq, CDIM, j0, kc * 32);
      c = wmma_f16(a, bf, c);
    }
    int n  = j0 + (ln & 15);
    int rb = (ln >> 4) * 8;
    float bias = qkv_b[n];
    if (jt < 8) {                                // Q (scaled)
#pragma unroll
      for (int r = 0; r < 8; ++r)
        qs[(mb + rb + r) * XSTR + n] = (_Float16)((c[r] + bias) * QSCALE);
    } else if (jt < 16) {                        // K
      int col = n - CDIM;
#pragma unroll
      for (int r = 0; r < 8; ++r)
        ks[(mb + rb + r) * XSTR + col] = (_Float16)(c[r] + bias);
    } else {                                     // V, channel-major (contiguous b128 store)
      int ch = n - 2 * CDIM;
#pragma unroll
      for (int r = 0; r < 8; ++r)
        vt[ch * VSTR + (mb + rb + r)] = (_Float16)(c[r] + bias);
    }
  }
  __syncthreads();

  // ---- per-head attention ----
  for (int h = 0; h < NHEAD; ++h) {
    const float* bh = biasH + (size_t)h * NN;

    // 2a: scores = q @ k^T (single K=32 WMMA per 16x16 tile) + rel bias + mask
    for (int t = wv; t < NT_M * NT_M; t += NWAVES) {
      int mt = t % NT_M, nt = t / NT_M;
      int mb = mt * 16, nb = nt * 16;
      v16h a  = load_a(qs, XSTR, mb, h * HD);
      v16h bf = load_b(ks, XSTR, nb, h * HD);
      v8f c = {};
      c = wmma_f16(a, bf, c);
      int n  = nb + (ln & 15);
      int rb = (ln >> 4) * 8;
      if (mt < 6 && nt < 6) {                    // interior: branch-free
#pragma unroll
        for (int r = 0; r < 8; ++r) {
          int m = mb + rb + r;
          int e = m * N_TOK + n;
          ss[m * SSTR + n] = c[r] + bh[e] + maskw[e];
        }
      } else {                                   // edge tiles: predicate padding
#pragma unroll
        for (int r = 0; r < 8; ++r) {
          int m = mb + rb + r;
          float val = -1e30f;
          if (m < N_TOK && n < N_TOK) {
            int e = m * N_TOK + n;
            val = c[r] + bh[e] + maskw[e];
          }
          ss[m * SSTR + n] = val;
        }
      }
    }
    __syncthreads();

    // 2b: softmax, one wave per row; lane owns column pairs -> b64 loads / packed b32 stores
    for (int row = wv; row < NP; row += NWAVES) {
      const float* srow = ss + row * SSTR;
      _Float16*    prow = ps + row * XSTR;
      if (row < N_TOK) {
        int c0 = 2 * ln;            // 0..63, always < 98
        int c2 = 64 + 2 * ln;       // 64..127
        float v0 = srow[c0];
        float v1 = srow[c0 + 1];
        float v2 = (c2     < N_TOK) ? srow[c2]     : -1e30f;
        float v3 = (c2 + 1 < N_TOK) ? srow[c2 + 1] : -1e30f;
        float mx = fmaxf(fmaxf(v0, v1), fmaxf(v2, v3));
#pragma unroll
        for (int off = 16; off > 0; off >>= 1) mx = fmaxf(mx, __shfl_xor(mx, off, 32));
        float e0 = __expf(v0 - mx);
        float e1 = __expf(v1 - mx);
        float e2 = (c2     < N_TOK) ? __expf(v2 - mx) : 0.f;
        float e3 = (c2 + 1 < N_TOK) ? __expf(v3 - mx) : 0.f;
        float sum = (e0 + e1) + (e2 + e3);
#pragma unroll
        for (int off = 16; off > 0; off >>= 1) sum += __shfl_xor(sum, off, 32);
        float inv = 1.f / sum;
        prow[c0]     = (_Float16)(e0 * inv);
        prow[c0 + 1] = (_Float16)(e1 * inv);
        prow[c2]     = (_Float16)(e2 * inv);
        prow[c2 + 1] = (_Float16)(e3 * inv);
      } else {
        prow[2 * ln]      = (_Float16)0.f;
        prow[2 * ln + 1]  = (_Float16)0.f;
        prow[64 + 2 * ln] = (_Float16)0.f;
        prow[65 + 2 * ln] = (_Float16)0.f;
      }
    }
    __syncthreads();

    // 2c: out_h = P @ V  (7 x 2 tiles, K = 4 x 32 over padded tokens)
    for (int t = wv; t < NT_M * 2; t += NWAVES) {
      int mt = t % NT_M, nt2 = t / NT_M;
      int mb = mt * 16;
      v8f c = {};
#pragma unroll
      for (int kc = 0; kc < 4; ++kc) {
        v16h a  = load_a(ps, XSTR, mb, kc * 32);
        v16h bf = load_b(vt, VSTR, h * HD + nt2 * 16, kc * 32);
        c = wmma_f16(a, bf, c);
      }
      int jj = h * HD + nt2 * 16 + (ln & 15);
      int rb = (ln >> 4) * 8;
#pragma unroll
      for (int r = 0; r < 8; ++r) os[(mb + rb + r) * XSTR + jj] = (_Float16)c[r];
    }
    __syncthreads();   // ss/ps reused by next head
  }

  // ---- stage 3: output projection, 7x8 tiles, write fp32 to HBM ----
  float* ob = out + (size_t)b * N_TOK * CDIM;
  for (int t = wv; t < NT_M * 8; t += NWAVES) {
    int mt = t % NT_M, jt = t / NT_M;
    int mb = mt * 16, j0 = jt * 16;
    v8f c = {};
#pragma unroll
    for (int kc = 0; kc < 4; ++kc) {
      v16h a  = load_a(os, XSTR, mb, kc * 32);
      v16h bf = load_b(wp, CDIM, j0, kc * 32);
      c = wmma_f16(a, bf, c);
    }
    int n  = j0 + (ln & 15);
    int rb = (ln >> 4) * 8;
    float bias = proj_b[n];
    if (mt < 6) {                                // interior rows: branch-free store
#pragma unroll
      for (int r = 0; r < 8; ++r) ob[(mb + rb + r) * CDIM + n] = c[r] + bias;
    } else {
#pragma unroll
      for (int r = 0; r < 8; ++r) {
        int m = mb + rb + r;
        if (m < N_TOK) ob[m * CDIM + n] = c[r] + bias;
      }
    }
  }
}

extern "C" void kernel_launch(void* const* d_in, const int* in_sizes, int n_in,
                              void* d_out, int out_size, void* d_ws, size_t ws_size,
                              hipStream_t stream) {
  (void)in_sizes; (void)n_in; (void)out_size; (void)ws_size;
  const float* x          = (const float*)d_in[0];
  const float* attn_mask  = (const float*)d_in[1];
  const float* qkv_w      = (const float*)d_in[2];
  const float* qkv_b      = (const float*)d_in[3];
  const float* proj_w     = (const float*)d_in[4];
  const float* proj_b     = (const float*)d_in[5];
  const float* bias_table = (const float*)d_in[6];
  const int*   rel_index  = (const int*)d_in[7];

  _Float16* wq    = (_Float16*)((char*)d_ws + WQ_OFF);
  _Float16* wp    = (_Float16*)((char*)d_ws + WP_OFF);
  float*    biasH = (float*)   ((char*)d_ws + BH_OFF);

  prep_weights<<<192, 256, 0, stream>>>(qkv_w, proj_w, wq, wp);
  prep_bias<<<(NHEAD * NN + 255) / 256, 256, 0, stream>>>(bias_table, rel_index, biasH);

  (void)hipFuncSetAttribute((const void*)win_attn3d_kernel,
                            hipFuncAttributeMaxDynamicSharedMemorySize, LDS_BYTES);

  win_attn3d_kernel<<<4096, NTHREADS, LDS_BYTES, stream>>>(
      x, attn_mask, qkv_b, proj_b, biasH, wq, wp, (float*)d_out);
}
